// EMLQuantizedLayer_42425686950159
// MI455X (gfx1250) — compile-verified, matrix operand order checked by reading, not run on hardware
//
#include <hip/hip_runtime.h>

// Ternary-quantized linear as bf16x2-split WMMA GEMM.
// M = 65536, K = 512, N = 512.
//
// Per block (256 thr = 8 wave32): stage 64 x-rows into LDS ONCE (async-to-LDS
// when available), split to hi/lo bf16 once (shared by all 8 waves), then
// K-loop feeds v_wmma_f32_16x16x32_bf16 from LDS. Weights are pre-quantized
// to exact ternary bf16 in d_ws (L2-resident, 512 KB).

typedef __attribute__((ext_vector_type(16))) __bf16 v16bf;
typedef __attribute__((ext_vector_type(8)))  __bf16 v8bf;
typedef __attribute__((ext_vector_type(4)))  __bf16 v4bf;
typedef __attribute__((ext_vector_type(8)))  float  v8f;
typedef __attribute__((ext_vector_type(4)))  float  v4f;
typedef __attribute__((ext_vector_type(4)))  int    v4i;

#define K_DIM  512
#define N_DIM  512
#define BM     64           // M-rows per block
#define RS     520          // bf16 row stride in LDS (+8 elems = 16B pad -> bank rotate)
#define RAWS   68           // f32 row stride for raw staging (+16B pad)
#define KC     64           // K chunk for async staging
#define THRESH 0.3f

#if defined(__has_builtin)
#if __has_builtin(__builtin_amdgcn_global_load_async_to_lds_b128)
#define HAVE_ASYNC_B128 1
#endif
#if __has_builtin(__builtin_amdgcn_s_wait_asynccnt)
#define HAVE_WAIT_ASYNC 1
#endif
#endif

// Typed, address-space-qualified pointers for the async-to-LDS builtin:
//   param0: v4i addrspace(1)* (non-const)   param1: v4i addrspace(3)*
typedef __attribute__((address_space(1))) v4i* gv4i_p;
typedef __attribute__((address_space(3))) v4i* lv4i_p;

union BF16x16 { v16bf v; v8bf h[2]; };

__device__ __forceinline__ void wait_async0() {
#if defined(HAVE_WAIT_ASYNC)
    __builtin_amdgcn_s_wait_asynccnt(0);
#else
    asm volatile("s_wait_asynccnt 0" ::: "memory");
#endif
}

__device__ __forceinline__ void split16(const float* fs, __bf16* hi, __bf16* lo) {
    #pragma unroll
    for (int e = 0; e < 16; ++e) {
        float  f = fs[e];
        __bf16 h = (__bf16)f;              // RNE hi part
        hi[e] = h;
        lo[e] = (__bf16)(f - (float)h);    // residual
    }
}

// ---------------- weight pre-quantization: f32 -> ternary bf16 ----------------
__global__ __launch_bounds__(256) void quantize_weights(const float* __restrict__ w,
                                                        __bf16* __restrict__ wq) {
    int i = (blockIdx.x * 256 + threadIdx.x) * 4;
    v4f f = *reinterpret_cast<const v4f*>(w + i);
    v4bf q;
    #pragma unroll
    for (int e = 0; e < 4; ++e) {
        float v = f[e];
        q[e] = (__bf16)((v > THRESH) ? 1.0f : ((v < -THRESH) ? -1.0f : 0.0f));
    }
    *reinterpret_cast<v4bf*>(wq + i) = q;
}

// ------------------------------- main GEMM -----------------------------------
template <bool PREQUANT>
__global__ __launch_bounds__(256) void ternary_wmma_gemm(
    const float*  __restrict__ x,    // [M, 512]
    const float*  __restrict__ w,    // [512, 512] f32 (fallback path)
    const __bf16* __restrict__ wq,   // [512, 512] ternary bf16 (prequant path)
    float*        __restrict__ out)  // [M, 512]
{
    extern __shared__ __align__(16) char smem[];
    __bf16* shi = (__bf16*)smem;                       // BM x RS
    __bf16* slo = (__bf16*)(smem + (size_t)BM * RS * 2);
#if defined(HAVE_ASYNC_B128)
    float*  sraw = (float*)(smem + 2 * (size_t)BM * RS * 2);  // BM x RAWS
#endif

    const int tid = threadIdx.x;
    const int m0  = blockIdx.x * BM;

    // ============ Stage x[m0..m0+63][0..511] -> hi/lo bf16 in LDS ============
#if defined(HAVE_ASYNC_B128)
    #pragma unroll 1
    for (int kc = 0; kc < K_DIM; kc += KC) {
        // async-copy BM x KC f32 tile (16 KB) into padded raw LDS
        #pragma unroll
        for (int q = 0; q < (BM * KC * 4) / (256 * 16); ++q) {  // 4 segs/thread
            int s   = tid + 256 * q;          // 16-byte segment id
            int row = s >> 4;                 // KC*4/16 = 16 segs per row
            int seg = s & 15;
            const float* gp = x + (size_t)(m0 + row) * K_DIM + kc + seg * 4;
            float*       lp = sraw + row * RAWS + seg * 4;
            __builtin_amdgcn_global_load_async_to_lds_b128((gv4i_p)gp, (lv4i_p)lp, 0, 0);
        }
        wait_async0();
        __syncthreads();
        // convert: 16 f32 per thread -> hi/lo bf16
        {
            int row = tid >> 2;
            int kl  = (tid & 3) * 16;
            __align__(16) float  fs[16];
            __align__(16) __bf16 h16[16], l16v[16];
            const float* rp = sraw + row * RAWS + kl;
            *reinterpret_cast<v4f*>(&fs[0])  = *reinterpret_cast<const v4f*>(rp + 0);
            *reinterpret_cast<v4f*>(&fs[4])  = *reinterpret_cast<const v4f*>(rp + 4);
            *reinterpret_cast<v4f*>(&fs[8])  = *reinterpret_cast<const v4f*>(rp + 8);
            *reinterpret_cast<v4f*>(&fs[12]) = *reinterpret_cast<const v4f*>(rp + 12);
            split16(fs, h16, l16v);
            __bf16* hp = shi + row * RS + kc + kl;
            __bf16* lp = slo + row * RS + kc + kl;
            *reinterpret_cast<v8bf*>(hp)     = *reinterpret_cast<v8bf*>(&h16[0]);
            *reinterpret_cast<v8bf*>(hp + 8) = *reinterpret_cast<v8bf*>(&h16[8]);
            *reinterpret_cast<v8bf*>(lp)     = *reinterpret_cast<v8bf*>(&l16v[0]);
            *reinterpret_cast<v8bf*>(lp + 8) = *reinterpret_cast<v8bf*>(&l16v[8]);
        }
        __syncthreads();   // raw buffer reusable after this
    }
#else
    // direct staging: global f32 -> registers -> hi/lo LDS
    #pragma unroll 1
    for (int q = 0; q < (BM * K_DIM) / (256 * 16); ++q) {       // 8 segs/thread
        int s   = tid + 256 * q;          // 16-float segment, 2048 total
        int row = s >> 5;                 // 512/16 = 32 segs per row
        int kl  = (s & 31) * 16;
        __align__(16) float  fs[16];
        __align__(16) __bf16 h16[16], l16v[16];
        const float* gp = x + (size_t)(m0 + row) * K_DIM + kl;
        *reinterpret_cast<v4f*>(&fs[0])  = *reinterpret_cast<const v4f*>(gp + 0);
        *reinterpret_cast<v4f*>(&fs[4])  = *reinterpret_cast<const v4f*>(gp + 4);
        *reinterpret_cast<v4f*>(&fs[8])  = *reinterpret_cast<const v4f*>(gp + 8);
        *reinterpret_cast<v4f*>(&fs[12]) = *reinterpret_cast<const v4f*>(gp + 12);
        split16(fs, h16, l16v);
        __bf16* hp = shi + row * RS + kl;
        __bf16* lp = slo + row * RS + kl;
        *reinterpret_cast<v8bf*>(hp)     = *reinterpret_cast<v8bf*>(&h16[0]);
        *reinterpret_cast<v8bf*>(hp + 8) = *reinterpret_cast<v8bf*>(&h16[8]);
        *reinterpret_cast<v8bf*>(lp)     = *reinterpret_cast<v8bf*>(&l16v[0]);
        *reinterpret_cast<v8bf*>(lp + 8) = *reinterpret_cast<v8bf*>(&l16v[8]);
    }
    __syncthreads();
#endif

    // ========================= WMMA main loop ================================
    const int wave  = tid >> 5;        // 0..7
    const int lane  = tid & 31;
    const int l16   = lane & 15;
    const int lhalf = lane >> 4;       // 0: K 0-7,16-23 ; 1: K 8-15,24-31
    const int n0    = wave * 64;

    v8f acc[4][4] = {};

    #pragma unroll 1
    for (int k0 = 0; k0 < K_DIM; k0 += 32) {
        BF16x16 ahi[4], alo[4], bq[4];

        #pragma unroll
        for (int i = 0; i < 4; ++i) {      // A frags from LDS (conflict-free)
            const __bf16* hp = shi + (size_t)(16 * i + l16) * RS + k0 + lhalf * 8;
            ahi[i].h[0] = *reinterpret_cast<const v8bf*>(hp);
            ahi[i].h[1] = *reinterpret_cast<const v8bf*>(hp + 16);
            const __bf16* lp = slo + (size_t)(16 * i + l16) * RS + k0 + lhalf * 8;
            alo[i].h[0] = *reinterpret_cast<const v8bf*>(lp);
            alo[i].h[1] = *reinterpret_cast<const v8bf*>(lp + 16);
        }

        #pragma unroll
        for (int j = 0; j < 4; ++j) {      // B frags (L2-hot weight)
            if (PREQUANT) {
                const __bf16* bp = wq + (size_t)(n0 + 16 * j + l16) * K_DIM + k0 + lhalf * 8;
                bq[j].h[0] = *reinterpret_cast<const v8bf*>(bp);
                bq[j].h[1] = *reinterpret_cast<const v8bf*>(bp + 16);
            } else {
                const float* wp = w + (size_t)(n0 + 16 * j + l16) * K_DIM + k0 + lhalf * 8;
                __align__(16) float fs[16];
                *reinterpret_cast<v4f*>(&fs[0])  = *reinterpret_cast<const v4f*>(wp + 0);
                *reinterpret_cast<v4f*>(&fs[4])  = *reinterpret_cast<const v4f*>(wp + 4);
                *reinterpret_cast<v4f*>(&fs[8])  = *reinterpret_cast<const v4f*>(wp + 16);
                *reinterpret_cast<v4f*>(&fs[12]) = *reinterpret_cast<const v4f*>(wp + 20);
                #pragma unroll
                for (int e = 0; e < 16; ++e) {
                    float f = fs[e];
                    bq[j].v[e] = (__bf16)((f > THRESH) ? 1.0f
                                       : ((f < -THRESH) ? -1.0f : 0.0f));
                }
            }
        }

        #pragma unroll
        for (int i = 0; i < 4; ++i) {
            #pragma unroll
            for (int j = 0; j < 4; ++j) {
                acc[i][j] = __builtin_amdgcn_wmma_f32_16x16x32_bf16(
                    false, ahi[i].v, false, bq[j].v, (short)0, acc[i][j], false, false);
                acc[i][j] = __builtin_amdgcn_wmma_f32_16x16x32_bf16(
                    false, alo[i].v, false, bq[j].v, (short)0, acc[i][j], false, false);
            }
        }
    }

    // ===== store: VGPR r -> row (r + 8*lhalf), col l16 =====
    #pragma unroll
    for (int i = 0; i < 4; ++i) {
        #pragma unroll
        for (int j = 0; j < 4; ++j) {
            float* op = out + (size_t)(m0 + 16 * i + lhalf * 8) * N_DIM
                            + (n0 + 16 * j + l16);
            #pragma unroll
            for (int r = 0; r < 8; ++r) {
                op[(size_t)r * N_DIM] = acc[i][j][r];
            }
        }
    }
}

extern "C" void kernel_launch(void* const* d_in, const int* in_sizes, int n_in,
                              void* d_out, int out_size, void* d_ws, size_t ws_size,
                              hipStream_t stream) {
    const float* x = (const float*)d_in[0];
    const float* w = (const float*)d_in[1];
    float* out     = (float*)d_out;

    const int M = in_sizes[0] / K_DIM;                 // 65536
    dim3 block(256);
    dim3 grid(M / BM);                                 // 1024 blocks

    // LDS: hi + lo (2 * 64*520*2 = 133120 B) + raw staging (64*68*4 = 17408 B)
    // Always reserve the max so host/device __has_builtin divergence is harmless.
    const size_t smemBytes = 2 * (size_t)BM * RS * 2 + (size_t)BM * RAWS * 4;

    const size_t wqBytes = (size_t)N_DIM * K_DIM * sizeof(__bf16);  // 512 KB
    if (ws_size >= wqBytes) {
        __bf16* wq = (__bf16*)d_ws;
        quantize_weights<<<dim3((N_DIM * K_DIM) / (256 * 4)), block, 0, stream>>>(w, wq);
        ternary_wmma_gemm<true><<<grid, block, smemBytes, stream>>>(x, w, wq, out);
    } else {
        ternary_wmma_gemm<false><<<grid, block, smemBytes, stream>>>(x, w, nullptr, out);
    }
}
